// DeformableConv2d_79645873537497
// MI455X (gfx1250) — matrix operand order repaired and checked by reading, not compiled, not run on hardware
//
#include <hip/hip_runtime.h>
#include <hip/hip_bf16.h>

typedef __attribute__((ext_vector_type(16))) _Float16 v16h;
typedef __attribute__((ext_vector_type(8)))  float    v8f;
typedef __attribute__((ext_vector_type(4)))  float    v4f;

constexpr int Bn  = 4;
constexpr int C   = 64;
constexpr int H   = 64;
constexpr int W   = 64;
constexpr int CO  = 64;
constexpr int KK  = 5;
constexpr int PAD = 2;
constexpr int P   = KK * KK;     // 25
constexpr int HW  = H * W;       // 4096
constexpr int NPIX = Bn * HW;    // 16384

// ---------------------------------------------------------------------------
// Kernel 0: x NCHW -> NHWC (channels contiguous for bilinear corner gathers)
// ---------------------------------------------------------------------------
__global__ void transpose_x_kernel(const float* __restrict__ x, float* __restrict__ xT) {
    int idx = blockIdx.x * blockDim.x + threadIdx.x;   // NPIX*C elements
    if (idx >= NPIX * C) return;
    int c   = idx % C;
    int pix = idx / C;
    int b   = pix / HW;
    int hw  = pix % HW;
    xT[idx] = x[((size_t)(b * C + c)) * HW + hw];      // coalesced writes
}

// ---------------------------------------------------------------------------
// Kernel 1: weight[CO][C][5][5] -> f16 A-tiles in WMMA register layout.
// A (16-bit, 16x32): lane l holds row M=l%16; halves j=0..15 ->
// K = j + (l>=16 ? 16 : 0) (contiguous, matching B).
// Stored flat as [p][ (cc*4+mt) ][lane][j]: one 8 KB block per p.
// ---------------------------------------------------------------------------
__global__ void prep_weights_kernel(const float* __restrict__ weight, _Float16* __restrict__ wA) {
    int idx = blockIdx.x * blockDim.x + threadIdx.x;   // P*2*4*32*16 = 102400
    if (idx >= P * 2 * 4 * 32 * 16) return;
    int j    = idx & 15;
    int lane = (idx >> 4) & 31;
    int mt   = (idx >> 9) & 3;
    int cc   = (idx >> 11) & 1;
    int p    = idx >> 12;
    int hi   = lane >> 4;
    int m    = lane & 15;
    int k    = j + (hi ? 16 : 0);
    int c    = cc * 32 + k;
    int o    = mt * 16 + m;
    wA[idx]  = (_Float16)weight[((size_t)o * C + c) * P + p];
}

// ---------------------------------------------------------------------------
// Kernel 2: offsets = pointwise( relu( depthwise3x3(x) + dw_b ) ) + pw_b
// Stores (dy,dx) pairs contiguously: offbuf[((b*P+p)*HW+hw)*2 + {0,1}]
// ---------------------------------------------------------------------------
__global__ void offsets_kernel(const float* __restrict__ x,
                               const float* __restrict__ dw_w, const float* __restrict__ dw_b,
                               const float* __restrict__ pw_w, const float* __restrict__ pw_b,
                               float* __restrict__ offbuf) {
    __shared__ float spw[2 * P * C];   // 50*64 = 12.8 KB
    __shared__ float sdw[C * 9];
    __shared__ float sdwb[C];
    for (int i = threadIdx.x; i < 2 * P * C; i += blockDim.x) spw[i] = pw_w[i];
    for (int i = threadIdx.x; i < C * 9;     i += blockDim.x) sdw[i] = dw_w[i];
    for (int i = threadIdx.x; i < C;         i += blockDim.x) sdwb[i] = dw_b[i];
    __syncthreads();

    int pix = blockIdx.x * blockDim.x + threadIdx.x;
    if (pix >= NPIX) return;
    int b  = pix / HW;
    int hw = pix % HW;
    int yy = hw / W;
    int xx = hw % W;

    float acc[2 * P];
#pragma unroll
    for (int q = 0; q < 2 * P; q++) acc[q] = pw_b[q];

    const float* xb = x + (size_t)b * C * HW;
    for (int c = 0; c < C; c++) {
        float h = sdwb[c];
#pragma unroll
        for (int i = 0; i < 3; i++) {
            int ys = yy + i - 1;
#pragma unroll
            for (int j = 0; j < 3; j++) {
                int xs = xx + j - 1;
                if (ys >= 0 && ys < H && xs >= 0 && xs < W)
                    h = fmaf(xb[c * HW + ys * W + xs], sdw[c * 9 + i * 3 + j], h);
            }
        }
        h = fmaxf(h, 0.0f);
#pragma unroll
        for (int q = 0; q < 2 * P; q++) acc[q] = fmaf(spw[q * C + c], h, acc[q]);
    }

    float* ob = offbuf + (size_t)b * P * HW * 2;
#pragma unroll
    for (int p = 0; p < P; p++) {
        ob[(p * HW + hw) * 2 + 0] = acc[2 * p];       // dy (channel 2p)
        ob[(p * HW + hw) * 2 + 1] = acc[2 * p + 1];   // dx (channel 2p+1)
    }
}

// ---------------------------------------------------------------------------
// CDNA5 async global->LDS copy (ASYNCcnt path). Per-lane 16-byte transfer:
//   global_load_async_to_lds_b128 vdst(LDS byte addr), vaddr(64b global), off
// Drained with s_wait_asynccnt 0 then a workgroup barrier.
// ---------------------------------------------------------------------------
__device__ __forceinline__ void async_copy_b128(unsigned int lds_byte_addr, const void* gaddr) {
    asm volatile("global_load_async_to_lds_b128 %0, %1, off"
                 :: "v"(lds_byte_addr), "v"(gaddr)
                 : "memory");
}
__device__ __forceinline__ void async_wait_all() {
    asm volatile("s_wait_asynccnt 0x0" ::: "memory");
}

// ---------------------------------------------------------------------------
// Kernel 3: fused bilinear gather + WMMA GEMM, A-tiles double-buffered in LDS.
// One wave = 16-pixel N-tile x all 64 output channels (4 f32 accumulators).
// Per p: stage next p's 8 KB of A-tiles via async DMA while computing
// 2 c-chunks x 4 WMMA on the current buffer.
// B (16-bit, 32x16): lane l = column N=l%16; halves j hold K=j+(l>=16?16:0).
// ---------------------------------------------------------------------------
__global__ void __launch_bounds__(256, 1)
deform_wmma_kernel(const float* __restrict__ xT, const _Float16* __restrict__ wA,
                   const float* __restrict__ offbuf, const float* __restrict__ bias,
                   float* __restrict__ out) {
    __shared__ _Float16 sA[2][4096];   // 2 x 8 KB: A-tiles for one p each

    int lane = threadIdx.x & 31;
    int wave = threadIdx.x >> 5;
    int tile = blockIdx.x * (blockDim.x >> 5) + wave;

    int npix0 = tile * 16;
    int b   = npix0 / HW;
    int hw0 = npix0 % HW;
    int col = lane & 15;
    int hi  = lane >> 4;           // 0: K 0-15, 1: K 16-31 of the chunk
    int hw  = hw0 + col;
    int yy  = hw / W;
    int xx  = hw % W;

    v8f acc[4];
#pragma unroll
    for (int mt = 0; mt < 4; mt++)
#pragma unroll
        for (int e = 0; e < 8; e++) acc[mt][e] = 0.0f;

    const float*  xb = xT + (size_t)b * HW * C;
    const float2* ob = (const float2*)offbuf + (size_t)b * P * HW;

    // ---- prologue: stage p=0 into buffer 0 (256 threads x 32 B = 8 KB) ----
    {
        const char*  src = (const char*)(wA);              // p = 0
        unsigned int dst = (unsigned int)(size_t)&sA[0][0];
        int t = threadIdx.x;
        async_copy_b128(dst + t * 32,      src + t * 32);
        async_copy_b128(dst + t * 32 + 16, src + t * 32 + 16);
        async_wait_all();
        __syncthreads();
    }

    int buf = 0;
    for (int p = 0; p < P; p++) {
        // ---- prefetch p+1 into the other buffer (overlaps with compute) ----
        if (p + 1 < P) {
            const char*  src = (const char*)(wA + (size_t)(p + 1) * 4096);
            unsigned int dst = (unsigned int)(size_t)&sA[buf ^ 1][0];
            int t = threadIdx.x;
            async_copy_b128(dst + t * 32,      src + t * 32);
            async_copy_b128(dst + t * 32 + 16, src + t * 32 + 16);
        }

        // ---- bilinear setup for this (pixel, p) ----
        float2 d  = ob[p * HW + hw];                       // (dy, dx)
        float  ys = (float)(yy + (p / 5) - PAD) + d.x;
        float  xs = (float)(xx + (p % 5) - PAD) + d.y;
        float  y0f = floorf(ys), x0f = floorf(xs);
        float  fy = ys - y0f,   fx = xs - x0f;
        int y0 = (int)y0f, x0 = (int)x0f;
        int y1 = y0 + 1,   x1 = x0 + 1;

        float v00 = (y0 >= 0 && y0 < H && x0 >= 0 && x0 < W) ? 1.0f : 0.0f;
        float v01 = (y0 >= 0 && y0 < H && x1 >= 0 && x1 < W) ? 1.0f : 0.0f;
        float v10 = (y1 >= 0 && y1 < H && x0 >= 0 && x0 < W) ? 1.0f : 0.0f;
        float v11 = (y1 >= 0 && y1 < H && x1 >= 0 && x1 < W) ? 1.0f : 0.0f;
        float w00 = (1.0f - fy) * (1.0f - fx) * v00;
        float w01 = (1.0f - fy) * fx          * v01;
        float w10 = fy          * (1.0f - fx) * v10;
        float w11 = fy          * fx          * v11;

        int y0c = min(max(y0, 0), H - 1), x0c = min(max(x0, 0), W - 1);
        int y1c = min(max(y1, 0), H - 1), x1c = min(max(x1, 0), W - 1);
        int base00 = (y0c * W + x0c) * C;
        int base01 = (y0c * W + x1c) * C;
        int base10 = (y1c * W + x0c) * C;
        int base11 = (y1c * W + x1c) * C;

        const v16h* sa = (const v16h*)&sA[buf][0];

#pragma unroll
        for (int cc = 0; cc < 2; cc++) {
            int c0 = cc * 32 + hi * 16;                    // 16 contiguous channels
            const v4f* p00 = (const v4f*)(xb + base00 + c0);
            const v4f* p01 = (const v4f*)(xb + base01 + c0);
            const v4f* p10 = (const v4f*)(xb + base10 + c0);
            const v4f* p11 = (const v4f*)(xb + base11 + c0);

            v16h bm;
#pragma unroll
            for (int q = 0; q < 4; q++) {
                v4f a0 = p00[q], a1 = p01[q], a2 = p10[q], a3 = p11[q];
#pragma unroll
                for (int e = 0; e < 4; e++) {
                    float v = w00 * a0[e] + w01 * a1[e] + w10 * a2[e] + w11 * a3[e];
                    bm[q * 4 + e] = (_Float16)v;
                }
            }

#pragma unroll
            for (int mt = 0; mt < 4; mt++) {
                v16h a = sa[(cc * 4 + mt) * 32 + lane];    // ds_load_b128 x2
                acc[mt] = __builtin_amdgcn_wmma_f32_16x16x32_f16(
                    false, a, false, bm, (short)0, acc[mt], false, false);
            }
        }

        // ---- drain this block's async stage of p+1, publish to all waves ----
        async_wait_all();
        __syncthreads();
        buf ^= 1;
    }

    // C/D layout (32-bit 16x16): VGPR r -> M = r + (lane>=16 ? 8 : 0), N = lane%16
    float* outb = out + (size_t)b * CO * HW;
#pragma unroll
    for (int mt = 0; mt < 4; mt++) {
#pragma unroll
        for (int r = 0; r < 8; r++) {
            int o = mt * 16 + r + hi * 8;
            outb[(size_t)o * HW + hw] = acc[mt][r] + bias[o];
        }
    }
}

// ---------------------------------------------------------------------------
extern "C" void kernel_launch(void* const* d_in, const int* in_sizes, int n_in,
                              void* d_out, int out_size, void* d_ws, size_t ws_size,
                              hipStream_t stream) {
    const float* x      = (const float*)d_in[0];
    const float* weight = (const float*)d_in[1];
    const float* bias   = (const float*)d_in[2];
    const float* dw_w   = (const float*)d_in[3];
    const float* dw_b   = (const float*)d_in[4];
    const float* pw_w   = (const float*)d_in[5];
    const float* pw_b   = (const float*)d_in[6];
    float* out = (float*)d_out;

    char* ws = (char*)d_ws;
    float*     xT     = (float*)ws;                                  // 4 MB
    _Float16*  wA     = (_Float16*)(ws + (4u << 20));                // 200 KB
    float*     offbuf = (float*)(ws + (4u << 20) + (256u << 10));    // 3.125 MB

    transpose_x_kernel<<<(NPIX * C + 255) / 256, 256, 0, stream>>>(x, xT);
    prep_weights_kernel<<<(P * 2 * 4 * 32 * 16 + 255) / 256, 256, 0, stream>>>(weight, wA);
    offsets_kernel<<<(NPIX + 255) / 256, 256, 0, stream>>>(x, dw_w, dw_b, pw_w, pw_b, offbuf);
    deform_wmma_kernel<<<(NPIX / 16) / 8, 256, 0, stream>>>(xT, wA, offbuf, bias, out);
}
